// GridBasedPooling_12283606468139
// MI455X (gfx1250) — matrix operand order repaired and checked by reading, not compiled, not run on hardware
//
#include <hip/hip_runtime.h>
#include <stdint.h>

typedef __attribute__((ext_vector_type(16))) __bf16 v16bf;
typedef __attribute__((ext_vector_type(8)))  __bf16 v8bf;
typedef __attribute__((ext_vector_type(8)))  float  v8f;

#define HID    128
#define NAG    64
#define G2C    64
#define KTOT   (G2C * HID)     /* 8192 */
#define CCH    4               /* cells per K-chunk */
#define KCH    (CCH * HID)     /* 512  */
#define NCHUNK (G2C / CCH)     /* 16   */
#define MROWS  32
#define SGSTRIDE (KCH + 8)     /* 520 bf16 -> bank-conflict-free A reads */
#define HSTRIDE  (HID + 8)     /* 136 bf16 */

/* Manual LDS arena (total 59392 B) so the fp32 async-load staging buffer can
 * alias sG (stage is fully consumed into bf16 sHid before sG is first written).
 *   [0      .. 33280)  sG   : bf16[MROWS][SGSTRIDE]   (alias: stage f32 32768B)
 *   [33280  .. 50688)  sHid : bf16[NAG][HSTRIDE]
 *   [50688  .. 54784)  sStart: u16[MROWS][G2C]
 *   [54784  .. 56832)  sCell : u8[MROWS][NAG]
 *   [56832  .. 58880)  sJl   : u8[MROWS][NAG]
 *   [58880  .. 59392)  sPos  : f32[NAG][2]
 */
#define SMEM_BYTES 59392
#define OFF_SHID   33280
#define OFF_START  50688
#define OFF_CELL   54784
#define OFF_JL     56832
#define OFF_POS    58880

// One-time conversion of W (fp32, [128, 8192]) to bf16 in workspace.
__global__ __launch_bounds__(256) void gbp_wconv(const float* __restrict__ W,
                                                 __bf16* __restrict__ Wb, int n) {
  int i = blockIdx.x * 256 + threadIdx.x;
  if (i < n) Wb[i] = (__bf16)W[i];
}

// Fused scatter (bucketed, race-free) + bf16 WMMA GEMM + bias.
// Workgroup = (batch b, half of 32 agents). 256 threads = 8 wave32.
__global__ __launch_bounds__(256) void gbp_main(
    const float*  __restrict__ hs,   // [64,64,128]
    const float*  __restrict__ pos,  // [64,64,2]
    const __bf16* __restrict__ Wb,   // [128,8192] bf16
    const float*  __restrict__ bias, // [128]
    float*        __restrict__ out)  // [64,64,128]
{
  __shared__ __align__(16) unsigned char smem[SMEM_BYTES];
  __bf16 (*sG)[SGSTRIDE]        = (__bf16(*)[SGSTRIDE])(smem);
  float*  stageF                = (float*)(smem);                 // aliases sG
  __bf16 (*sHid)[HSTRIDE]       = (__bf16(*)[HSTRIDE])(smem + OFF_SHID);
  unsigned short (*sStart)[G2C] = (unsigned short(*)[G2C])(smem + OFF_START);
  unsigned char (*sCell)[NAG]   = (unsigned char(*)[NAG])(smem + OFF_CELL);
  unsigned char (*sJl)[NAG]     = (unsigned char(*)[NAG])(smem + OFF_JL);
  float (*sPos)[2]              = (float(*)[2])(smem + OFF_POS);

  const int tid   = threadIdx.x;
  const int lane  = tid & 31;
  const int wave  = tid >> 5;
  const int b     = blockIdx.x >> 1;
  const int ibase = (blockIdx.x & 1) * MROWS;

  if (tid < NAG) {
    sPos[tid][0] = pos[(b * NAG + tid) * 2 + 0];
    sPos[tid][1] = pos[(b * NAG + tid) * 2 + 1];
  }

  // Kick off async DMA of hidden[b] (fp32, 32KB) straight into the LDS stage
  // (GLOBAL_LOAD_ASYNC_TO_LDS_B128, tracked by ASYNCcnt) — overlaps with the
  // cell-index math below.
  const float* hsb = hs + (size_t)b * NAG * HID;
#pragma unroll
  for (int e4 = tid; e4 < NAG * HID / 4; e4 += 256) {
    unsigned ldsoff = (unsigned)(uintptr_t)(stageF + e4 * 4);
    unsigned long long ga = (unsigned long long)(uintptr_t)(hsb + e4 * 4);
    asm volatile("global_load_async_to_lds_b128 %0, %1, off"
                 :: "v"(ldsoff), "v"(ga) : "memory");
  }

  __syncthreads();   // sPos ready

  // Relative-position grid-cell indices (trunc-toward-zero, clamp, matches ref).
  for (int e = tid; e < MROWS * NAG; e += 256) {
    int il = e >> 6, j = e & 63;
    int ig = ibase + il;
    float rx = sPos[j][0] - sPos[ig][0];
    float ry = sPos[j][1] - sPos[ig][1];
    int gx = (int)((rx + 2.0f) * 2.0f);   // (rel + NEIGH/2) / CELL
    int gy = (int)((ry + 2.0f) * 2.0f);
    gx = gx < 0 ? 0 : (gx > 7 ? 7 : gx);
    gy = gy < 0 ? 0 : (gy > 7 ? 7 : gy);
    sCell[il][j] = (j == ig) ? (unsigned char)255 : (unsigned char)(gx * 8 + gy);
  }

  asm volatile("s_wait_asynccnt 0x0" ::: "memory");  // stage resident in LDS
  __syncthreads();

  // Stage (f32) -> sHid (bf16); frees the sG region before the chunk loop.
  for (int e = tid; e < NAG * HID; e += 256)
    sHid[e >> 7][e & 127] = (__bf16)stageF[e];

  // Per-agent counting sort of neighbors by cell (one thread per agent row).
  if (tid < MROWS) {
    const int il = tid;
    for (int c = 0; c < G2C; ++c) sStart[il][c] = 0;
    for (int j = 0; j < NAG; ++j) {
      int c = sCell[il][j];
      if (c != 255) sStart[il][c]++;
    }
    unsigned s = 0;
    for (int c = 0; c < G2C; ++c) {
      unsigned t = sStart[il][c];
      sStart[il][c] = (unsigned short)s;
      s += t;
    }
    for (int j = 0; j < NAG; ++j) {
      int c = sCell[il][j];
      if (c != 255) {
        unsigned p = sStart[il][c]++;
        sJl[il][p] = (unsigned char)j;
      }
    }
    // sStart[il][c] now holds the END of bucket c; start(c) = end(c-1).
  }
  __syncthreads();

  // WMMA tiling: wave -> (row tile mt in {0,1}, two 16-col tiles nt0, nt0+1).
  const int mt    = wave >> 2;
  const int nt0   = (wave & 3) * 2;
  const int M     = lane & 15;
  const int koffA = (lane & 16) ? 16 : 0;  // A k order used here: hi lanes K+16 (contiguous)
  const int koffB = (lane & 16) ? 16 : 0;  // B k order used here (same as A): hi lanes K+16
  const __bf16* wrow0 = Wb + (size_t)(nt0 * 16 + M) * KTOT + koffB;
  const __bf16* wrow1 = wrow0 + (size_t)16 * KTOT;
  v8f acc0 = {}; v8f acc1 = {};

  const int p1_il = tid >> 3;          // owned agent row
  const int p1_h0 = (tid & 7) * 16;    // owned 16-wide h slice

  for (int ch = 0; ch < NCHUNK; ++ch) {
    // ---- Phase 1: build bf16 grid slice for cells [ch*CCH, ch*CCH+CCH) ----
    {
      float accv[CCH][16];
#pragma unroll
      for (int c = 0; c < CCH; ++c)
#pragma unroll
        for (int h = 0; h < 16; ++h) accv[c][h] = 0.0f;
#pragma unroll
      for (int cc = 0; cc < CCH; ++cc) {
        const int cell = ch * CCH + cc;
        const int lo = cell ? (int)sStart[p1_il][cell - 1] : 0;
        const int hi = (int)sStart[p1_il][cell];
        for (int p = lo; p < hi; ++p) {
          const int j = sJl[p1_il][p];
#pragma unroll
          for (int h = 0; h < 16; ++h)
            accv[cc][h] += (float)sHid[j][p1_h0 + h];
        }
#pragma unroll
        for (int h = 0; h < 16; ++h)
          sG[p1_il][cc * HID + p1_h0 + h] = (__bf16)accv[cc][h];
      }
    }
    __syncthreads();

    // ---- Phase 2: 16 K-steps of v_wmma_f32_16x16x32_bf16 over this chunk ----
    const __bf16* sArow = &sG[mt * 16 + M][0];
#pragma unroll
    for (int ks = 0; ks < KCH / 32; ++ks) {
      const int kb = ks * 32;
      union { v16bf v; v8bf h[2]; } a;
      a.h[0] = *(const v8bf*)(sArow + kb + koffA);        // K = kb+{0..7}(+16)
      a.h[1] = *(const v8bf*)(sArow + kb + 8 + koffA);    // K = kb+{8..15}(+16)
      v16bf b0 = *(const v16bf*)(wrow0 + (size_t)ch * KCH + kb);
      v16bf b1 = *(const v16bf*)(wrow1 + (size_t)ch * KCH + kb);
      acc0 = __builtin_amdgcn_wmma_f32_16x16x32_bf16(false, a.v, false, b0,
                                                     (short)0, acc0, false, false);
      acc1 = __builtin_amdgcn_wmma_f32_16x16x32_bf16(false, a.v, false, b1,
                                                     (short)0, acc1, false, false);
    }
    if (ch + 1 < NCHUNK) {
      __builtin_prefetch(wrow0 + (size_t)(ch + 1) * KCH, 0, 1);
      __builtin_prefetch(wrow1 + (size_t)(ch + 1) * KCH, 0, 1);
    }
    __syncthreads();
  }

  // ---- Epilogue: C/D layout (VGPR r: M=r lanes0-15, M=r+8 lanes16-31) ----
  const int Mhi   = (lane & 16) ? 8 : 0;
  const int ncol  = lane & 15;
  const int rowg0 = b * NAG + ibase + mt * 16 + Mhi;
  const float bia0 = bias[nt0 * 16 + ncol];
  const float bia1 = bias[nt0 * 16 + 16 + ncol];
#pragma unroll
  for (int r = 0; r < 8; ++r) {
    const int og = (rowg0 + r) * HID;
    out[og + nt0 * 16 + ncol]       = acc0[r] + bia0;
    out[og + (nt0 + 1) * 16 + ncol] = acc1[r] + bia1;
  }
}

extern "C" void kernel_launch(void* const* d_in, const int* in_sizes, int n_in,
                              void* d_out, int out_size, void* d_ws, size_t ws_size,
                              hipStream_t stream) {
  const float* hs   = (const float*)d_in[0];  // hidden_states [64,64,128]
  const float* pos  = (const float*)d_in[1];  // positions     [64,64,2]
  const float* W    = (const float*)d_in[2];  // W             [128,8192]
  const float* bia  = (const float*)d_in[3];  // b             [128]
  float* out = (float*)d_out;
  __bf16* Wb = (__bf16*)d_ws;                 // 2 MB bf16 copy of W

  const int n = HID * KTOT;                   // 1,048,576
  gbp_wconv<<<(n + 255) / 256, 256, 0, stream>>>(W, Wb, n);
  gbp_main<<<dim3(NAG * 2), dim3(256), 0, stream>>>(hs, pos, Wb, bia, out);
}
